// FragmentMatchingGNN_2705829396656
// MI455X (gfx1250) — matrix operand order repaired and checked by reading, not compiled, not run on hardware
//
#include <hip/hip_runtime.h>
#include <hip/hip_bf16.h>

#define N_NODES 50000
#define E_EDGES 800000
#define ETOT    850000   // E + N self loops
#define GRAPHS  64
#define HDIM    128

typedef __attribute__((ext_vector_type(16))) _Float16 v16h;
typedef __attribute__((ext_vector_type(8)))  float    v8f;

// ---- ordered-uint encoding for exact float atomic max ----
__device__ __forceinline__ unsigned fkey(float f) {
  unsigned b = __float_as_uint(f);
  return (b & 0x80000000u) ? ~b : (b | 0x80000000u);
}
__device__ __forceinline__ float fdecode(unsigned k) {
  unsigned b = (k & 0x80000000u) ? (k & 0x7FFFFFFFu) : ~k;
  return __uint_as_float(b);
}
__device__ __forceinline__ void atomAddF(float* p, float v) {
  __hip_atomic_fetch_add(p, v, __ATOMIC_RELAXED, __HIP_MEMORY_SCOPE_AGENT);
}

// =====================================================================
// g = hin @ W   (hin: [N,K] f32, W: [K,128] f32, g: [N,128] f32)
// One wave per 16x16 output tile; block = 256 threads = 8 waves covering
// all 128 output columns of a 16-row strip. K in {64,128}, chunks of 32.
// Uses v_wmma_f32_16x16x32_f16 (f32 accumulate).
// =====================================================================
__global__ void gat_gemm_wmma(const float* __restrict__ hin,
                              const float* __restrict__ W,
                              float* __restrict__ g, int K) {
  const int wave = threadIdx.x >> 5;
  const int lane = threadIdx.x & 31;
  const int lhi  = lane >> 4;          // 0 or 1 (lane half)
  const int m0   = blockIdx.x * 16;
  const int n0   = wave * 16;
  const int Mrow = m0 + (lane & 15);
  const int Ncol = n0 + (lane & 15);

  v8f acc = {};
  for (int k0 = 0; k0 < K; k0 += 32) {
    if (k0 + 32 < K)
      __builtin_prefetch(hin + (size_t)Mrow * K + k0 + 32, 0, 1); // global_prefetch_b8
    v16h a, b;
    // A fragment (ISA 7.12.2, 16-bit A 16x32): lane holds row M=lane&15,
    // element e -> K = (e&7) | ((e>>3)<<4) | (lhi<<3)
    const float* arow = hin + (size_t)Mrow * K;
#pragma unroll
    for (int e = 0; e < 16; e += 2) {
      int kk = k0 + (e & 7) + ((e >> 3) << 4) + (lhi << 3);
      float2 av = *(const float2*)(arow + kk);
      a[e]     = (_Float16)av.x;
      a[e + 1] = (_Float16)av.y;
    }
    // B fragment (32x16, K rows x N cols): lane holds col N=lane&15,
// element e -> K = (e&7) | ((e>>3)<<4) | (lhi<<3)
#pragma unroll
    for (int e = 0; e < 16; e++) {
      int kk = k0 + (e & 7) + ((e >> 3) << 4) + (lhi << 3);
      b[e] = (_Float16)W[(size_t)kk * HDIM + Ncol];
    }
    acc = __builtin_amdgcn_wmma_f32_16x16x32_f16(
        false, a, false, b, (short)0, acc, false, false);
  }
  // C/D layout: VGPR r -> M = r + 8*lhi, N = lane&15
#pragma unroll
  for (int r = 0; r < 8; r++) {
    int Mr = m0 + r + lhi * 8;
    g[(size_t)Mr * HDIM + Ncol] = acc[r];
  }
}

// per-(node,head) attention logits: a_s = <g[n,h,:], att_src>, a_d likewise
__global__ void attn_scores_kernel(const float* __restrict__ g,
                                   const float* __restrict__ asv,
                                   const float* __restrict__ adv,
                                   float* __restrict__ as_, float* __restrict__ ad_) {
  int t = blockIdx.x * blockDim.x + threadIdx.x;
  if (t >= N_NODES * 4) return;
  int n = t >> 2, hh = t & 3;
  const float* gp = g + (size_t)n * HDIM + hh * 32;
  const float* sv = asv + hh * 32;
  const float* dv = adv + hh * 32;
  float s = 0.f, d = 0.f;
#pragma unroll 8
  for (int c = 0; c < 32; c++) { float gv = gp[c]; s += gv * sv[c]; d += gv * dv[c]; }
  as_[t] = s; ad_[t] = d;
}

__device__ __forceinline__ void edge_sd(const int* ei, int e, int& s, int& d) {
  if (e < E_EDGES) { s = ei[e]; d = ei[E_EDGES + e]; }
  else             { s = d = e - E_EDGES; }   // self loops
}

__global__ void edge_max_kernel(const int* __restrict__ ei,
                                const float* __restrict__ as_,
                                const float* __restrict__ ad_,
                                unsigned* __restrict__ mkey) {
  int t = blockIdx.x * blockDim.x + threadIdx.x;
  if (t >= ETOT * 4) return;
  int e = t >> 2, hh = t & 3, s, d;
  edge_sd(ei, e, s, d);
  float v = as_[s * 4 + hh] + ad_[d * 4 + hh];
  v = v > 0.f ? v : 0.2f * v;                  // leaky_relu(0.2)
  atomicMax(&mkey[d * 4 + hh], fkey(v));
}

__global__ void edge_expsum_kernel(const int* __restrict__ ei,
                                   const float* __restrict__ as_,
                                   const float* __restrict__ ad_,
                                   const unsigned* __restrict__ mkey,
                                   float* __restrict__ ssum,
                                   float* __restrict__ pbuf) {
  int t = blockIdx.x * blockDim.x + threadIdx.x;
  if (t >= ETOT * 4) return;
  int e = t >> 2, hh = t & 3, s, d;
  edge_sd(ei, e, s, d);
  float v = as_[s * 4 + hh] + ad_[d * 4 + hh];
  v = v > 0.f ? v : 0.2f * v;
  float p = expf(v - fdecode(mkey[d * 4 + hh]));
  pbuf[t] = p;
  atomAddF(&ssum[d * 4 + hh], p);
}

__global__ void edge_agg_kernel(const int* __restrict__ ei,
                                const float* __restrict__ g,
                                const float* __restrict__ pbuf,
                                const float* __restrict__ ssum,
                                float* __restrict__ outacc) {
  long long t = (long long)blockIdx.x * blockDim.x + threadIdx.x;
  if (t >= (long long)ETOT * HDIM) return;
  int e = (int)(t >> 7), ch = (int)(t & 127), hh = ch >> 5, s, d;
  edge_sd(ei, e, s, d);
  float alpha = pbuf[e * 4 + hh] / ssum[d * 4 + hh];
  atomAddF(&outacc[(size_t)d * HDIM + ch], g[(size_t)s * HDIM + ch] * alpha);
}

__global__ void bn_relu_kernel(const float* __restrict__ acc,
                               const float* __restrict__ bias,
                               const float* __restrict__ gamma,
                               const float* __restrict__ beta,
                               const float* __restrict__ mean,
                               const float* __restrict__ var,
                               float* __restrict__ hout) {
  int t = blockIdx.x * blockDim.x + threadIdx.x;
  if (t >= N_NODES * HDIM) return;
  int ch = t & 127;
  float v = acc[t] + bias[ch];
  v = (v - mean[ch]) * rsqrtf(var[ch] + 1e-5f) * gamma[ch] + beta[ch];
  hout[t] = fmaxf(v, 0.f);
}

// attention-pool logit per node: relu(h@W1+b1)@W2 + b2 ; block=64 thr/node
__global__ void pool_logits_kernel(const float* __restrict__ h,
                                   const float* __restrict__ W1,
                                   const float* __restrict__ b1,
                                   const float* __restrict__ W2,
                                   const float* __restrict__ b2,
                                   float* __restrict__ logits) {
  int n = blockIdx.x, j = threadIdx.x;
  const float* hn = h + (size_t)n * HDIM;
  float acc = b1[j];
#pragma unroll 8
  for (int k = 0; k < HDIM; k++) acc += hn[k] * W1[k * 64 + j];
  __shared__ float red[64];
  red[j] = fmaxf(acc, 0.f) * W2[j];
  __syncthreads();
  for (int s = 32; s > 0; s >>= 1) { if (j < s) red[j] += red[j + s]; __syncthreads(); }
  if (j == 0) logits[n] = red[0] + b2[0];
}

__global__ void frag_max_kernel(const float* __restrict__ logits,
                                const int* __restrict__ frag,
                                unsigned* __restrict__ fmax) {
  int n = blockIdx.x * blockDim.x + threadIdx.x;
  if (n >= N_NODES) return;
  atomicMax(&fmax[frag[n]], fkey(logits[n]));
}

__global__ void frag_sum_kernel(const float* __restrict__ logits,
                                const int* __restrict__ frag,
                                const unsigned* __restrict__ fmax,
                                float* __restrict__ fsum) {
  int n = blockIdx.x * blockDim.x + threadIdx.x;
  if (n >= N_NODES) return;
  int f = frag[n];
  atomAddF(&fsum[f], expf(logits[n] - fdecode(fmax[f])));
}

__global__ void pool_accum_kernel(const float* __restrict__ h,
                                  const float* __restrict__ logits,
                                  const int* __restrict__ frag,
                                  const int* __restrict__ batch,
                                  const unsigned* __restrict__ fmax,
                                  const float* __restrict__ fsum,
                                  float* __restrict__ fragpool) {
  long long t = (long long)blockIdx.x * blockDim.x + threadIdx.x;
  if (t >= (long long)N_NODES * HDIM) return;
  int n = (int)(t >> 7), ch = (int)(t & 127);
  int f = frag[n];
  float w = expf(logits[n] - fdecode(fmax[f])) / fsum[f];
  atomAddF(&fragpool[(size_t)f * GRAPHS * HDIM + (size_t)batch[n] * HDIM + ch],
           h[t] * w);
}

// fused graph-level heads: one block (128 thr) per graph
__global__ void head_kernel(const float* __restrict__ fragpool,
    const float* iW1, const float* ib1, const float* iW2, const float* ib2,
    const float* cW1, const float* cb1, const float* cW2, const float* cb2,
    const float* clng, const float* clnb,
    const float* mW1, const float* mb1, const float* mW2, const float* mb2,
    const float* clsW, const float* clsb, const float* regW, const float* regb,
    const float* mcW, const float* mcb, float* __restrict__ out) {
  int g = blockIdx.x, t = threadIdx.x;   // 128 threads
  __shared__ float comb[256], t1[128], inter[64], zh[64], red[128], sh1[32], sh2[16];
  comb[t]       = fragpool[(size_t)g * HDIM + t];
  comb[128 + t] = fragpool[(size_t)GRAPHS * HDIM + (size_t)g * HDIM + t];
  __syncthreads();
  // interaction MLP
  float a = ib1[t];
  for (int k = 0; k < 256; k++) a += comb[k] * iW1[k * 128 + t];
  t1[t] = fmaxf(a, 0.f);
  __syncthreads();
  if (t < 64) {
    float b = ib2[t];
    for (int k = 0; k < 128; k++) b += t1[k] * iW2[k * 64 + t];
    inter[t] = fmaxf(b, 0.f);
  }
  __syncthreads();
  // contrastive projector for f1 then f2
  for (int f = 0; f < 2; f++) {
    const float* emb = comb + f * 128;
    if (t < 64) {
      float b = cb1[t];
      for (int k = 0; k < 128; k++) b += emb[k] * cW1[k * 64 + t];
      zh[t] = fmaxf(b, 0.f);
    }
    __syncthreads();
    float zz = cb2[t];
    for (int k = 0; k < 64; k++) zz += zh[k] * cW2[k * 128 + t];
    red[t] = zz; __syncthreads();
    for (int s = 64; s > 0; s >>= 1) { if (t < s) red[t] += red[t + s]; __syncthreads(); }
    float mu = red[0] / 128.f; __syncthreads();
    red[t] = (zz - mu) * (zz - mu); __syncthreads();
    for (int s = 64; s > 0; s >>= 1) { if (t < s) red[t] += red[t + s]; __syncthreads(); }
    float var = red[0] / 128.f; __syncthreads();
    float zn = (zz - mu) * rsqrtf(var + 1e-5f) * clng[t] + clnb[t];
    red[t] = zn * zn; __syncthreads();
    for (int s = 64; s > 0; s >>= 1) { if (t < s) red[t] += red[t + s]; __syncthreads(); }
    float nrm = fmaxf(sqrtf(red[0]), 1e-12f); __syncthreads();
    out[(size_t)f * GRAPHS * 128 + (size_t)g * 128 + t] = zn / nrm;
  }
  // multitask head
  if (t < 32) {
    float b = mb1[t];
    for (int k = 0; k < 64; k++) b += inter[k] * mW1[k * 32 + t];
    sh1[t] = fmaxf(b, 0.f);
  }
  __syncthreads();
  if (t < 16) {
    float b = mb2[t];
    for (int k = 0; k < 32; k++) b += sh1[k] * mW2[k * 16 + t];
    sh2[t] = fmaxf(b, 0.f);
  }
  __syncthreads();
  const size_t base = 2 * (size_t)GRAPHS * 128;
  if (t == 0) {
    float c = clsb[0], r = regb[0];
    for (int k = 0; k < 16; k++) { c += sh2[k] * clsW[k]; r += sh2[k] * regW[k]; }
    out[base + g] = c;
    out[base + GRAPHS + g] = 1.f / (1.f + expf(-r));
  }
  if (t < 2) {
    float m = mcb[t];
    for (int k = 0; k < 16; k++) m += sh2[k] * mcW[k * 2 + t];
    out[base + 2 * GRAPHS + (size_t)g * 2 + t] = m;
  }
}

extern "C" void kernel_launch(void* const* d_in, const int* in_sizes, int n_in,
                              void* d_out, int out_size, void* d_ws, size_t ws_size,
                              hipStream_t stream) {
  const float* x        = (const float*)d_in[0];
  const int*   ei       = (const int*)d_in[1];
  const int*   batch    = (const int*)d_in[2];
  const int*   frag     = (const int*)d_in[3];
  const float* gat_W0   = (const float*)d_in[4];
  const float* gat_W12  = (const float*)d_in[5];
  const float* att_src  = (const float*)d_in[6];
  const float* att_dst  = (const float*)d_in[7];
  const float* gat_b    = (const float*)d_in[8];
  const float* bn_gamma = (const float*)d_in[9];
  const float* bn_beta  = (const float*)d_in[10];
  const float* bn_mean  = (const float*)d_in[11];
  const float* bn_var   = (const float*)d_in[12];
  const float* pool_W1  = (const float*)d_in[13];
  const float* pool_b1  = (const float*)d_in[14];
  const float* pool_W2  = (const float*)d_in[15];
  const float* pool_b2  = (const float*)d_in[16];
  const float* inter_W1 = (const float*)d_in[17];
  const float* inter_b1 = (const float*)d_in[18];
  const float* inter_W2 = (const float*)d_in[19];
  const float* inter_b2 = (const float*)d_in[20];
  const float* c_W1     = (const float*)d_in[21];
  const float* c_b1     = (const float*)d_in[22];
  const float* c_W2     = (const float*)d_in[23];
  const float* c_b2     = (const float*)d_in[24];
  const float* c_ln_g   = (const float*)d_in[25];
  const float* c_ln_b   = (const float*)d_in[26];
  const float* mt_W1    = (const float*)d_in[27];
  const float* mt_b1    = (const float*)d_in[28];
  const float* mt_W2    = (const float*)d_in[29];
  const float* mt_b2    = (const float*)d_in[30];
  const float* cls_W    = (const float*)d_in[31];
  const float* cls_b    = (const float*)d_in[32];
  const float* reg_W    = (const float*)d_in[33];
  const float* reg_b    = (const float*)d_in[34];
  const float* mc_W     = (const float*)d_in[35];
  const float* mc_b     = (const float*)d_in[36];

  // workspace carve-up (floats)
  char* base = (char*)d_ws;
  size_t off = 0;
  auto alloc = [&](size_t nfloats) { void* p = base + off; off += nfloats * 4; return p; };
  float*    h        = (float*)alloc((size_t)N_NODES * HDIM);
  float*    gbuf     = (float*)alloc((size_t)N_NODES * HDIM);
  float*    outacc   = (float*)alloc((size_t)N_NODES * HDIM);
  float*    as_      = (float*)alloc((size_t)N_NODES * 4);
  float*    ad_      = (float*)alloc((size_t)N_NODES * 4);
  unsigned* mkey     = (unsigned*)alloc((size_t)N_NODES * 4);
  float*    ssum     = (float*)alloc((size_t)N_NODES * 4);
  float*    pbuf     = (float*)alloc((size_t)ETOT * 4);
  float*    logits   = (float*)alloc((size_t)N_NODES);
  float*    fragpool = (float*)alloc((size_t)2 * GRAPHS * HDIM);
  unsigned* fmaxkey  = (unsigned*)alloc(2);
  float*    fsum     = (float*)alloc(2);

  const int THR = 256;
  const int nodesHeads = (N_NODES * 4 + THR - 1) / THR;
  const int edgesHeads = (ETOT * 4 + THR - 1) / THR;
  const int edgesChan  = (int)(((long long)ETOT * HDIM + THR - 1) / THR);
  const int nodesChan  = (N_NODES * HDIM + THR - 1) / THR;

  for (int i = 0; i < 3; i++) {
    const float* hin = (i == 0) ? x : h;
    const int K = (i == 0) ? 64 : 128;
    const float* W = (i == 0) ? gat_W0 : gat_W12 + (size_t)(i - 1) * 128 * 128;

    gat_gemm_wmma<<<N_NODES / 16, 256, 0, stream>>>(hin, W, gbuf, K);
    attn_scores_kernel<<<nodesHeads, THR, 0, stream>>>(
        gbuf, att_src + i * 128, att_dst + i * 128, as_, ad_);
    hipMemsetAsync(mkey, 0, (size_t)N_NODES * 4 * sizeof(unsigned), stream);
    hipMemsetAsync(ssum, 0, (size_t)N_NODES * 4 * sizeof(float), stream);
    hipMemsetAsync(outacc, 0, (size_t)N_NODES * HDIM * sizeof(float), stream);
    edge_max_kernel<<<edgesHeads, THR, 0, stream>>>(ei, as_, ad_, mkey);
    edge_expsum_kernel<<<edgesHeads, THR, 0, stream>>>(ei, as_, ad_, mkey, ssum, pbuf);
    edge_agg_kernel<<<edgesChan, THR, 0, stream>>>(ei, gbuf, pbuf, ssum, outacc);
    bn_relu_kernel<<<nodesChan, THR, 0, stream>>>(
        outacc, gat_b + i * 128, bn_gamma + i * 128, bn_beta + i * 128,
        bn_mean + i * 128, bn_var + i * 128, h);
  }

  pool_logits_kernel<<<N_NODES, 64, 0, stream>>>(h, pool_W1, pool_b1, pool_W2,
                                                 pool_b2, logits);
  hipMemsetAsync(fmaxkey, 0, 2 * sizeof(unsigned), stream);
  hipMemsetAsync(fsum, 0, 2 * sizeof(float), stream);
  hipMemsetAsync(fragpool, 0, (size_t)2 * GRAPHS * HDIM * sizeof(float), stream);
  frag_max_kernel<<<(N_NODES + THR - 1) / THR, THR, 0, stream>>>(logits, frag, fmaxkey);
  frag_sum_kernel<<<(N_NODES + THR - 1) / THR, THR, 0, stream>>>(logits, frag, fmaxkey, fsum);
  pool_accum_kernel<<<nodesChan, THR, 0, stream>>>(h, logits, frag, batch, fmaxkey,
                                                   fsum, fragpool);
  head_kernel<<<GRAPHS, 128, 0, stream>>>(
      fragpool, inter_W1, inter_b1, inter_W2, inter_b2,
      c_W1, c_b1, c_W2, c_b2, c_ln_g, c_ln_b,
      mt_W1, mt_b1, mt_W2, mt_b2,
      cls_W, cls_b, reg_W, reg_b, mc_W, mc_b, (float*)d_out);
}
